// MultiHeadAttentionBlock_72198400245972
// MI455X (gfx1250) — matrix operand order repaired and checked by reading, not compiled, not run on hardware
//
#include <hip/hip_runtime.h>
#include <hip/hip_bf16.h>

typedef __attribute__((ext_vector_type(16))) _Float16 v16h;
typedef __attribute__((ext_vector_type(8)))  float    v8f;

#define DMODEL 1024
#define SEQ    2048
#define NH     16
#define DK     64
#define NB     2
#define MTOT   (NB * SEQ)   // 4096

// ---------------- fragment loaders (CDNA5 WMMA wave32 layouts) ----------------

// A-matrix 16x32 f16, source row-major tile[m*ld + k]
__device__ __forceinline__ v16h a_frag_load(const _Float16* __restrict__ tile,
                                            int ld, int lane) {
  const int m  = lane & 15;
  const int kh = lane >> 4;
  const _Float16* rp = tile + m * ld;
  union { v16h v; _Float16 e[16]; } u;
#pragma unroll
  for (int vr = 0; vr < 8; ++vr) {
    const int k = kh * 16 + vr * 2;
    u.e[2 * vr]     = rp[k];
    u.e[2 * vr + 1] = rp[k + 1];
  }
  return u.v;
}

// B-matrix 32x16 f16, source holds B^T row-major: tile[n*ld + k]
__device__ __forceinline__ v16h b_frag_load_nk(const _Float16* __restrict__ tile,
                                               int ld, int lane) {
  const int n  = lane & 15;
  const int kh = lane >> 4;
  const _Float16* rp = tile + n * ld;
  union { v16h v; _Float16 e[16]; } u;
#pragma unroll
  for (int vr = 0; vr < 8; ++vr) {
    const int k = kh * 16 + vr * 2;
    u.e[2 * vr]     = rp[k];
    u.e[2 * vr + 1] = rp[k + 1];
  }
  return u.v;
}

// B-matrix 32x16 f16, source holds B row-major: tile[k*kstride + n]
__device__ __forceinline__ v16h b_frag_load_kn(const _Float16* __restrict__ tile,
                                               int kstride, int lane) {
  const int n  = lane & 15;
  const int kh = lane >> 4;
  union { v16h v; _Float16 e[16]; } u;
#pragma unroll
  for (int vr = 0; vr < 8; ++vr) {
    const int k = kh * 16 + vr * 2;
    u.e[2 * vr]     = tile[(size_t)k * kstride + n];
    u.e[2 * vr + 1] = tile[(size_t)(k + 1) * kstride + n];
  }
  return u.v;
}

__device__ __forceinline__ v8f wmma_f16(v16h a, v16h b, v8f c) {
  return __builtin_amdgcn_wmma_f32_16x16x32_f16(false, a, false, b,
                                                (short)0, c, false, false);
}

__device__ __forceinline__ v8f v8f_zero() {
  v8f z = {0.f, 0.f, 0.f, 0.f, 0.f, 0.f, 0.f, 0.f};
  return z;
}

// ---------------- TDM (Tensor Data Mover) tile staging ----------------
#if __has_builtin(__builtin_amdgcn_tensor_load_to_lds)
#define USE_TDM 1
#else
#define USE_TDM 0
#endif

#if USE_TDM
typedef __attribute__((ext_vector_type(4))) unsigned int u32x4;
typedef __attribute__((ext_vector_type(8))) int          i32x8;
typedef __attribute__((ext_vector_type(4))) int          i32x4;

__device__ __forceinline__ unsigned lds_offset_of(const void* p) {
  return (unsigned)(unsigned long long)
         (__attribute__((address_space(3))) const void*)p;
}

// Contiguous 8 KB global->LDS copy as a 1D TDM tensor tile:
// data_size=8B, tile_dim0 = tensor_dim0 = tensor_dim0_stride = 1024 elements.
__device__ __forceinline__ void tdm_copy_8kb(void* lds_dst, const void* gsrc) {
  const unsigned long long ga = (unsigned long long)gsrc;
  u32x4 g0;
  g0[0] = 1u;                                        // count=1 (valid user D#)
  g0[1] = lds_offset_of(lds_dst);                    // lds_addr
  g0[2] = (unsigned)(ga & 0xffffffffu);              // global_addr[31:0]
  g0[3] = (unsigned)((ga >> 32) & 0x01ffffffu)       // global_addr[56:32]
          | (2u << 30);                              // type = 2 ("image")
  i32x8 g1;
  g1[0] = 0x00030000;          // workgroup_mask=0, data_size=3 (8B), no flags
  g1[1] = (int)(1024u << 16);  // atomic_barrier_addr=0 | tensor_dim0.lo16=1024
  g1[2] = (int)(1u << 16);     // tensor_dim0.hi16=0   | tensor_dim1=1
  g1[3] = (int)(1024u << 16);  // tensor_dim1.hi16=0   | tile_dim0=1024
  g1[4] = 1;                   // tile_dim1=1 | tile_dim2=0
  g1[5] = 1024;                // tensor_dim0_stride[31:0]
  g1[6] = 0;                   // stride0.hi16 | tensor_dim1_stride.lo16
  g1[7] = 0;                   // tensor_dim1_stride.hi32
  i32x4 g2 = {1, 1, 0, 0};         // tensor_dim2=1, tensor_dim3=1
  i32x4 g3 = {0, (int)(1u << 16), 0, 0};  // dim3_stride=0, tensor_dim4=1
#if __clang_major__ >= 23
  i32x8 z8 = {0, 0, 0, 0, 0, 0, 0, 0};
  __builtin_amdgcn_tensor_load_to_lds(g0, g1, g2, g3, z8, 0);
#else
  __builtin_amdgcn_tensor_load_to_lds(g0, g1, g2, g3, 0);
#endif
}
#endif  // USE_TDM

// Stage one 64-key K tile and V tile (8 KB each, contiguous since ld == DK).
__device__ __forceinline__ void stage_tiles(_Float16* kdst, _Float16* vdst,
                                            const _Float16* ksrc,
                                            const _Float16* vsrc,
                                            int tid, int wave) {
#if USE_TDM
  if (wave == 0) {            // one wave drives the DMA engine; EXEC ignored
    tdm_copy_8kb(kdst, ksrc);
    tdm_copy_8kb(vdst, vsrc);
  }
  (void)tid;
#else
  const uint4* ks = (const uint4*)ksrc;
  const uint4* vs = (const uint4*)vsrc;
  uint4* kd = (uint4*)kdst;
  uint4* vd = (uint4*)vdst;
  kd[tid] = ks[tid]; kd[tid + 256] = ks[tid + 256];
  vd[tid] = vs[tid]; vd[tid + 256] = vs[tid + 256];
  (void)wave;
#endif
}

__device__ __forceinline__ void wait_tiles(int wave) {
#if USE_TDM
  if (wave == 0) __builtin_amdgcn_s_wait_tensorcnt(0);
#else
  (void)wave;  // sync path: __syncthreads() after this is sufficient
#endif
}

// ---------------- GEMM: C[M,N] = A[M,K] @ W[N,K]^T + bias ----------------
// M=4096, N=K=1024. Block: 256 threads (8 waves), tile 128x64, KC=32.
// OUT_MODE 0: f16 output scattered to head-split [B,H,S,DK]
// OUT_MODE 1: f32 output row-major [M,N]
template <int OUT_MODE>
__global__ __launch_bounds__(256) void proj_gemm(const float* __restrict__ A,
                                                 const float* __restrict__ W,
                                                 const float* __restrict__ bias,
                                                 void* __restrict__ outp) {
  constexpr int BM = 128, BN = 64, KC = 32, K = 1024;
  __shared__ _Float16 As[BM * KC];  // 8 KB
  __shared__ _Float16 Bs[BN * KC];  // 4 KB
  const int tid = threadIdx.x, lane = tid & 31, wave = tid >> 5;
  const int m0 = blockIdx.x * BM, n0 = blockIdx.y * BN;

  v8f acc[4] = {v8f_zero(), v8f_zero(), v8f_zero(), v8f_zero()};

  for (int k0 = 0; k0 < K; k0 += KC) {
    __syncthreads();
    // stage A 128x32 fp32 -> f16 : 1024 float4 units, 4 per thread
#pragma unroll
    for (int i = 0; i < 4; ++i) {
      const int idx = tid + i * 256;
      const int r = idx >> 3, c = (idx & 7) * 4;
      const float4 f = *(const float4*)(A + (size_t)(m0 + r) * K + k0 + c);
      _Float16* d = As + r * KC + c;
      d[0] = (_Float16)f.x; d[1] = (_Float16)f.y;
      d[2] = (_Float16)f.z; d[3] = (_Float16)f.w;
    }
    // stage W 64x32 fp32 -> f16 : 512 float4 units, 2 per thread
#pragma unroll
    for (int i = 0; i < 2; ++i) {
      const int idx = tid + i * 256;
      const int r = idx >> 3, c = (idx & 7) * 4;
      const float4 f = *(const float4*)(W + (size_t)(n0 + r) * K + k0 + c);
      _Float16* d = Bs + r * KC + c;
      d[0] = (_Float16)f.x; d[1] = (_Float16)f.y;
      d[2] = (_Float16)f.z; d[3] = (_Float16)f.w;
    }
    __syncthreads();
    const v16h a = a_frag_load(As + wave * 16 * KC, KC, lane);
#pragma unroll
    for (int nt = 0; nt < 4; ++nt) {
      const v16h b = b_frag_load_nk(Bs + nt * 16 * KC, KC, lane);
      acc[nt] = wmma_f16(a, b, acc[nt]);
    }
  }

  const int kh = lane >> 4, nn = lane & 15;
#pragma unroll
  for (int nt = 0; nt < 4; ++nt) {
    const int gn = n0 + nt * 16 + nn;
    const float bv = bias[gn];
#pragma unroll
    for (int r = 0; r < 8; ++r) {
      const int gm = m0 + wave * 16 + r + kh * 8;
      const float val = acc[nt][r] + bv;
      if (OUT_MODE == 0) {
        const int b = gm >> 11, s = gm & (SEQ - 1);
        const int h = gn >> 6, d = gn & (DK - 1);
        ((_Float16*)outp)[((size_t)(b * NH + h) * SEQ + s) * DK + d] = (_Float16)val;
      } else {
        ((float*)outp)[(size_t)gm * DMODEL + gn] = val;
      }
    }
  }
}

// ---------------- Flash attention ----------------
// Grid: (B*H, SEQ/128), 256 threads (8 waves). Each wave owns 16 q rows.
// K/V tiles double-buffered in LDS, staged by the Tensor Data Mover.
__global__ __launch_bounds__(256) void flash_attn(const _Float16* __restrict__ Q,
                                                  const _Float16* __restrict__ K,
                                                  const _Float16* __restrict__ V,
                                                  const int* __restrict__ mask,
                                                  float* __restrict__ ctx) {
  __shared__ _Float16 Kt[2][64 * DK];   // 2 x 8 KB
  __shared__ _Float16 Vt[2][64 * DK];   // 2 x 8 KB
  __shared__ _Float16 Pt[8][16 * 32];   // 8 KB (per-wave P staging)

  const int tid = threadIdx.x, lane = tid & 31, wave = tid >> 5;
  const int bh = blockIdx.x;                       // b*NH + h
  const int q0 = blockIdx.y * 128 + wave * 16;     // wave's first q row
  const int khalf = lane >> 4, nn = lane & 15;
  const size_t head = (size_t)bh * SEQ;

  // Q fragments for this wave: 16 rows x 64 dims (two 16x32 chunks)
  const _Float16* Qp = Q + (head + q0) * DK;
  const v16h qa0 = a_frag_load(Qp, DK, lane);
  const v16h qa1 = a_frag_load(Qp + 32, DK, lane);

  v8f o[4] = {v8f_zero(), v8f_zero(), v8f_zero(), v8f_zero()};
  float mi[8], li[8];
#pragma unroll
  for (int r = 0; r < 8; ++r) { mi[r] = -1e30f; li[r] = 0.f; }

  // prologue: kick off DMA of the first K/V tile
  stage_tiles(Kt[0], Vt[0], K + head * DK, V + head * DK, tid, wave);

  for (int kb = 0; kb < SEQ; kb += 64) {
    const int cur = (kb >> 6) & 1;
    wait_tiles(wave);        // TENSORcnt -> 0 on the issuing wave
    __syncthreads();         // publish tile to all waves; WAR fence for reuse
    if (kb + 64 < SEQ)       // overlap next tile's DMA with compute
      stage_tiles(Kt[cur ^ 1], Vt[cur ^ 1],
                  K + (head + kb + 64) * DK, V + (head + kb + 64) * DK,
                  tid, wave);
    const _Float16* Kc = Kt[cur];
    const _Float16* Vc = Vt[cur];

#pragma unroll 1
    for (int kt = 0; kt < 64; kt += 32) {
      // scores: two 16x16 tiles over 32 keys, contraction over DK=64
      v8f s0 = v8f_zero(), s1 = v8f_zero();
      s0 = wmma_f16(qa0, b_frag_load_nk(Kc + (kt +  0) * DK,      DK, lane), s0);
      s0 = wmma_f16(qa1, b_frag_load_nk(Kc + (kt +  0) * DK + 32, DK, lane), s0);
      s1 = wmma_f16(qa0, b_frag_load_nk(Kc + (kt + 16) * DK,      DK, lane), s1);
      s1 = wmma_f16(qa1, b_frag_load_nk(Kc + (kt + 16) * DK + 32, DK, lane), s1);

      const int key0 = kb + kt + nn;
      const int key1 = key0 + 16;
      float p0[8], p1[8], alpha[8];
#pragma unroll
      for (int r = 0; r < 8; ++r) {
        const int gq = q0 + r + khalf * 8;
        float sa = s0[r] * 0.125f;   // 1/sqrt(DK)
        float sb = s1[r] * 0.125f;
        if (mask[(size_t)gq * SEQ + key0] == 0) sa = -1e9f;
        if (mask[(size_t)gq * SEQ + key1] == 0) sb = -1e9f;
        // row max across the 16 lanes that hold this row's 32 keys
        float mx = fmaxf(sa, sb);
#pragma unroll
        for (int off = 8; off >= 1; off >>= 1)
          mx = fmaxf(mx, __shfl_xor(mx, off, 32));
        const float mnew = fmaxf(mi[r], mx);
        alpha[r] = __expf(mi[r] - mnew);
        mi[r] = mnew;
        p0[r] = __expf(sa - mnew);
        p1[r] = __expf(sb - mnew);
        float rs = p0[r] + p1[r];
#pragma unroll
        for (int off = 8; off >= 1; off >>= 1)
          rs += __shfl_xor(rs, off, 32);
        li[r] = li[r] * alpha[r] + rs;
      }

      // rescale O, stash P (16x32 f16) into this wave's LDS region
      _Float16* pw = Pt[wave];
#pragma unroll
      for (int r = 0; r < 8; ++r) {
        const int m = r + khalf * 8;
        pw[m * 32 + nn]      = (_Float16)p0[r];
        pw[m * 32 + 16 + nn] = (_Float16)p1[r];
#pragma unroll
        for (int g = 0; g < 4; ++g) o[g][r] *= alpha[r];
      }
      // intra-wave LDS store->load: DS ops are in-order per wave; wait explicitly
      asm volatile("s_wait_dscnt 0" ::: "memory");
      const v16h pa = a_frag_load(pw, 32, lane);
#pragma unroll
      for (int g = 0; g < 4; ++g)
        o[g] = wmma_f16(pa, b_frag_load_kn(Vc + kt * DK + g * 16, DK, lane), o[g]);
    }
  }

  // epilogue: normalize and write ctx as [B, S, D] fp32
  const int b = bh / NH, h = bh % NH;
#pragma unroll
  for (int r = 0; r < 8; ++r) {
    const int gq = q0 + r + khalf * 8;
    const float inv = 1.0f / li[r];
    float* dst = ctx + ((size_t)b * SEQ + gq) * DMODEL + h * DK;
#pragma unroll
    for (int g = 0; g < 4; ++g) dst[g * 16 + nn] = o[g][r] * inv;
  }
}

// ---------------- launch ----------------
extern "C" void kernel_launch(void* const* d_in, const int* in_sizes, int n_in,
                              void* d_out, int out_size, void* d_ws, size_t ws_size,
                              hipStream_t stream) {
  (void)in_sizes; (void)n_in; (void)out_size; (void)ws_size;
  const float* q    = (const float*)d_in[0];
  const float* k    = (const float*)d_in[1];
  const float* v    = (const float*)d_in[2];
  const int*   mask = (const int*)d_in[3];
  const float* w_q  = (const float*)d_in[4];
  const float* b_q  = (const float*)d_in[5];
  const float* w_k  = (const float*)d_in[6];
  const float* b_k  = (const float*)d_in[7];
  const float* w_v  = (const float*)d_in[8];
  const float* b_v  = (const float*)d_in[9];
  const float* w_o  = (const float*)d_in[10];
  const float* b_o  = (const float*)d_in[11];

  // workspace: Qp/Kp/Vp f16 head-split (8 MB each) + ctx fp32 (16 MB) = 40 MB
  const size_t nProj = (size_t)MTOT * DMODEL;
  _Float16* Qp = (_Float16*)d_ws;
  _Float16* Kp = Qp + nProj;
  _Float16* Vp = Kp + nProj;
  float*    ctx = (float*)(Vp + nProj);

  const dim3 gg(MTOT / 128, DMODEL / 64);
  proj_gemm<0><<<gg, 256, 0, stream>>>(q, w_q, b_q, (void*)Qp);
  proj_gemm<0><<<gg, 256, 0, stream>>>(k, w_k, b_k, (void*)Kp);
  proj_gemm<0><<<gg, 256, 0, stream>>>(v, w_v, b_v, (void*)Vp);
  flash_attn<<<dim3(NB * NH, SEQ / 128), 256, 0, stream>>>(Qp, Kp, Vp, mask, ctx);
  proj_gemm<1><<<gg, 256, 0, stream>>>(ctx, w_o, b_o, d_out);
}
